// SelfAttention_v1_5128190951599
// MI455X (gfx1250) — matrix operand order repaired and checked by reading, not compiled, not run on hardware
//
#include <hip/hip_runtime.h>
#include <cmath>

// Self-attention, S=8192, D=768, single head.
// Strategy: f16 WMMA (16x16x32, f32 accum) everywhere; all intermediates
// (Q, K, V) stored in per-lane WMMA fragment layouts (contiguous k = 16h+e, not ISA table order) so
// every operand load is one contiguous 32B v16h load. Attention is
// flash-style (online softmax), never materializing the 8192x8192 matrix.
// Workspace: 4*S*D + 3*D*D f16 elements (~54 MB).

#define SQ   8192
#define DIM  768
#define KT   (DIM/32)   // 24 k-tiles of 32
#define NTD  (DIM/16)   // 48 n-tiles of 16
#define MTS  (SQ/16)    // 512 row tiles

typedef __attribute__((ext_vector_type(16))) _Float16 v16h;
typedef __attribute__((ext_vector_type(8)))  float    v8f;

static __device__ __forceinline__ v8f wmma16(v16h a, v16h b, v8f c) {
  // D = A(16x32 f16) x B(32x16 f16) + C(16x16 f32)
  return __builtin_amdgcn_wmma_f32_16x16x32_f16(false, a, false, b,
                                                (short)0, c, false, false);
}

// ---- pack X (f32 row-major) into WMMA A-fragment layout (f16) ------------
// A pack (contiguous k) 16x32 f16: lane<16 row=lane, K=0..15; lane>=16 row=lane-16,
// K=16..31. Element e (0..15): k_local = e + (lane<16?0:16).
__global__ void pack_x_kernel(const float* __restrict__ X,
                              _Float16* __restrict__ XA) {
  unsigned idx  = blockIdx.x * 256u + threadIdx.x;
  unsigned e    = idx & 15u;
  unsigned lane = (idx >> 4) & 31u;
  unsigned tile = idx >> 9;
  unsigned mt = tile / KT, kt = tile % KT;
  unsigned m = mt * 16u + (lane & 15u);
  unsigned k = kt * 32u + e + (lane < 16u ? 0u : 16u);
  XA[idx] = (_Float16)X[(size_t)m * DIM + k];
}

// ---- pack W (f32 [n][k]) into WMMA B-fragment layout for y = x @ W^T -----
// B pack (contiguous k) 32x16 f16: lane<16 col=lane, K=e; lane>=16 col=lane-16, K=16+e.
__global__ void pack_w_kernel(const float* __restrict__ W,
                              _Float16* __restrict__ WB) {
  unsigned idx  = blockIdx.x * 256u + threadIdx.x;
  unsigned e    = idx & 15u;
  unsigned lane = (idx >> 4) & 31u;
  unsigned tile = idx >> 9;
  unsigned nt = tile / KT, kt = tile % KT;
  unsigned n = nt * 16u + (lane & 15u);
  unsigned k = kt * 32u + e + (lane < 16u ? 0u : 16u);
  WB[idx] = (_Float16)W[(size_t)n * DIM + k];
}

// ---- projection GEMM: [16x16 tile per wave], D scattered into the layout
// needed by the attention kernel. mode: 0=Q(A-pack) 1=K(B-pack) 2=V(B-pack).
__global__ void __launch_bounds__(256)
proj_kernel(const _Float16* __restrict__ XA, const _Float16* __restrict__ WB,
            _Float16* __restrict__ OUT, int mode) {
  __shared__ alignas(32) _Float16 xs[KT * 512];
  const unsigned tid  = threadIdx.x;
  const unsigned mt   = blockIdx.x / 6u;
  const unsigned ng   = blockIdx.x % 6u;
  const unsigned w    = tid >> 5, lane = tid & 31u, nl = lane & 15u;
  const unsigned nt   = ng * 8u + w;
  const unsigned base = (lane < 16u) ? 0u : 8u;

  { // stage one 16-row X block (all 24 k-tiles, contiguous in XA)
    const uint4* src = (const uint4*)(XA + (size_t)mt * KT * 512);
    uint4* dst = (uint4*)xs;
    for (unsigned i = tid; i < (KT * 512u * 2u) / 16u; i += 256u) dst[i] = src[i];
  }
  __syncthreads();

  v8f acc = {};
  #pragma unroll 4
  for (unsigned kt = 0; kt < KT; ++kt) {
    v16h a = *(const v16h*)(xs + (kt * 32u + lane) * 16u);
    v16h b = *(const v16h*)(WB + ((size_t)(nt * KT + kt) * 32u + lane) * 16u);
    acc = wmma16(a, b, acc);
  }

  // D layout: lane<16 -> col=lane, row=r ; lane>=16 -> col=lane-16, row=8+r
  if (mode == 0) {            // Q -> A-pack for scores (tile mt, kt2=nt/2)
    unsigned e   = nl;
    unsigned hif = (nt & 1u) * 16u;
    size_t tb = (size_t)(mt * KT + (nt >> 1)) * 32u;
    #pragma unroll
    for (unsigned r = 0; r < 8; ++r)
      OUT[(tb + base + r + hif) * 16u + e] = (_Float16)acc[r];
  } else if (mode == 1) {     // K -> B-pack for scores (seq-tile=mt, kt2=nt/2)
    size_t tb = (size_t)(mt * KT + (nt >> 1)) * 32u + (nt & 1u) * 16u;
    #pragma unroll
    for (unsigned r = 0; r < 8; ++r)
      OUT[(tb + base + r) * 16u + nl] = (_Float16)acc[r];
  } else {                    // V -> B-pack for P@V (feat-tile=nt, kt=mt/2)
    size_t tb = ((size_t)(nt * (SQ / 32) + (mt >> 1)) * 32u
                 + nl + (mt & 1u) * 16u) * 16u;
    #pragma unroll
    for (unsigned r = 0; r < 8; ++r)
      OUT[tb + base + r] = (_Float16)acc[r];
  }
}

// ---- flash attention: one workgroup (8 waves) per 16 query rows ----------
__global__ void __launch_bounds__(256)
attn_kernel(const _Float16* __restrict__ QA, const _Float16* __restrict__ KB,
            const _Float16* __restrict__ VB, float* __restrict__ OUT) {
  __shared__ alignas(32) _Float16 qs[KT * 512];  // Q fragments, 24 KB
  __shared__ alignas(32) _Float16 ps[4 * 512];   // P fragments (16x128), 4 KB
  __shared__ float wmaxs[8 * 16], wsums[8 * 16];
  __shared__ float mrow[16], lrow[16], mnew[16], arow[16];

  const unsigned tid  = threadIdx.x;
  const unsigned mt   = blockIdx.x;
  const unsigned w    = tid >> 5, lane = tid & 31u, nl = lane & 15u;
  const unsigned base = (lane < 16u) ? 0u : 8u;

  { // stage Q block (contiguous in QA)
    const uint4* src = (const uint4*)(QA + (size_t)mt * KT * 512);
    uint4* dst = (uint4*)qs;
    for (unsigned i = tid; i < (KT * 512u * 2u) / 16u; i += 256u) dst[i] = src[i];
  }
  if (tid < 16u) { mrow[tid] = -INFINITY; lrow[tid] = 0.f; }
  __syncthreads();

  v8f o[6] = {};                        // 16x96 f32 output slice per wave
  const float sc = 0.036084391824351615f;      // 1/sqrt(768)
  const unsigned pe   = nl;   // P A-pack element
  const unsigned phif = (w & 1u) * 16u;
  const unsigned pktp = w >> 1;

  for (unsigned j = 0; j < SQ / 128u; ++j) {
    // ---- scores: 16x16 tile, columns j*128 + w*16 .. +16
    const _Float16* kb = KB + (size_t)(j * 8u + w) * KT * 512u;
    if (j + 1u < SQ / 128u)
      __builtin_prefetch(KB + (size_t)((j + 1u) * 8u + w) * KT * 512u, 0, 1);
    v8f s = {};
    #pragma unroll 4
    for (unsigned kt = 0; kt < KT; ++kt) {
      v16h a = *(const v16h*)(qs + (kt * 32u + lane) * 16u);
      v16h b = *(const v16h*)(kb + (kt * 32u + lane) * 16u);
      s = wmma16(a, b, s);
    }
    float sv[8], rmax[8];
    #pragma unroll
    for (int r = 0; r < 8; ++r) { sv[r] = s[r] * sc; rmax[r] = sv[r]; }
    // per-row max over 16 columns (row r lives in reg r across a 16-lane group)
    #pragma unroll
    for (int off = 1; off < 16; off <<= 1)
      #pragma unroll
      for (int r = 0; r < 8; ++r)
        rmax[r] = fmaxf(rmax[r], __shfl_xor(rmax[r], off));
    if (nl == 0u) {
      #pragma unroll
      for (int r = 0; r < 8; ++r) wmaxs[w * 16u + base + r] = rmax[r];
    }
    __syncthreads();                                   // (A)
    if (tid < 16u) {
      float mb = wmaxs[tid];
      #pragma unroll
      for (int ww = 1; ww < 8; ++ww) mb = fmaxf(mb, wmaxs[ww * 16 + tid]);
      float mo = mrow[tid], mn = fmaxf(mo, mb);
      mnew[tid] = mn;
      arow[tid] = __expf(mo - mn);                     // 0 on first block
    }
    __syncthreads();                                   // (B)
    float av[8], rsum[8];
    #pragma unroll
    for (int r = 0; r < 8; ++r) {
      float pv = __expf(sv[r] - mnew[base + r]);
      rsum[r] = pv;
      ps[(pktp * 32u + base + r + phif) * 16u + pe] = (_Float16)pv;
      av[r] = arow[base + r];
    }
    #pragma unroll
    for (int nt = 0; nt < 6; ++nt)
      #pragma unroll
      for (int r = 0; r < 8; ++r) o[nt][r] *= av[r];   // rescale accumulator
    #pragma unroll
    for (int off = 1; off < 16; off <<= 1)
      #pragma unroll
      for (int r = 0; r < 8; ++r) rsum[r] += __shfl_xor(rsum[r], off);
    if (nl == 0u) {
      #pragma unroll
      for (int r = 0; r < 8; ++r) wsums[w * 16u + base + r] = rsum[r];
    }
    __syncthreads();                                   // (C)
    if (tid < 16u) {
      float ss = wsums[tid];
      #pragma unroll
      for (int ww = 1; ww < 8; ++ww) ss += wsums[ww * 16 + tid];
      lrow[tid] = lrow[tid] * arow[tid] + ss;
      mrow[tid] = mnew[tid];
    }
    // ---- O += P(16x128) @ V(128x96 slice), 24 WMMAs per wave
    #pragma unroll
    for (unsigned ktp = 0; ktp < 4; ++ktp) {
      v16h pa = *(const v16h*)(ps + (ktp * 32u + lane) * 16u);
      #pragma unroll
      for (unsigned nt = 0; nt < 6; ++nt) {
        v16h vb = *(const v16h*)(VB +
            ((size_t)((w * 6u + nt) * (SQ / 32) + (j * 4u + ktp)) * 32u + lane) * 16u);
        o[nt] = wmma16(pa, vb, o[nt]);
      }
    }
    // next iteration's (A)+(B) barriers protect ps reuse
  }
  __syncthreads();
  float lv[8];
  #pragma unroll
  for (int r = 0; r < 8; ++r) lv[r] = 1.0f / lrow[base + r];
  #pragma unroll
  for (unsigned nt = 0; nt < 6; ++nt)
    #pragma unroll
    for (unsigned r = 0; r < 8; ++r)
      OUT[(size_t)(mt * 16u + base + r) * DIM + w * 96u + nt * 16u + nl] =
          o[nt][r] * lv[r];
}

extern "C" void kernel_launch(void* const* d_in, const int* in_sizes, int n_in,
                              void* d_out, int out_size, void* d_ws, size_t ws_size,
                              hipStream_t stream) {
  const float* X  = (const float*)d_in[0];
  const float* Wq = (const float*)d_in[1];
  const float* Wk = (const float*)d_in[2];
  const float* Wv = (const float*)d_in[3];
  float* OUT = (float*)d_out;

  _Float16* ws = (_Float16*)d_ws;
  const size_t SD = (size_t)SQ * DIM;   // 6,291,456
  const size_t DD = (size_t)DIM * DIM;  //   589,824
  _Float16* XA = ws;                    // X, A-fragment layout
  _Float16* QA = ws + SD;               // Q, A-fragment layout (scores)
  _Float16* KBp = ws + 2 * SD;          // K, B-fragment layout (scores)
  _Float16* VBp = ws + 3 * SD;          // V, B-fragment layout (P@V)
  _Float16* WQ = ws + 4 * SD;
  _Float16* WK = WQ + DD;
  _Float16* WV = WK + DD;               // total ~54 MB of workspace

  pack_x_kernel<<<(unsigned)(SD / 256), 256, 0, stream>>>(X, XA);
  pack_w_kernel<<<(unsigned)(DD / 256), 256, 0, stream>>>(Wq, WQ);
  pack_w_kernel<<<(unsigned)(DD / 256), 256, 0, stream>>>(Wk, WK);
  pack_w_kernel<<<(unsigned)(DD / 256), 256, 0, stream>>>(Wv, WV);

  proj_kernel<<<MTS * 6, 256, 0, stream>>>(XA, WQ, QA, 0);
  proj_kernel<<<MTS * 6, 256, 0, stream>>>(XA, WK, KBp, 1);
  proj_kernel<<<MTS * 6, 256, 0, stream>>>(XA, WV, VBp, 2);

  attn_kernel<<<MTS, 256, 0, stream>>>(QA, KBp, VBp, OUT);
}
